// PixelBlock_58755152609940
// MI455X (gfx1250) — matrix operand order repaired and checked by reading, not compiled, not run on hardware
//
#include <hip/hip_runtime.h>
#include <hip/hip_bf16.h>

typedef __attribute__((ext_vector_type(8)))  _Float16 v8h;
typedef __attribute__((ext_vector_type(16))) _Float16 v16h;
typedef __attribute__((ext_vector_type(8)))  float    v8f;

#define BB     2
#define LL     2304      // 48*48
#define CIN    192
#define EMB    256
#define NH     8
#define HD     32
#define NEG_BIG (-1.0e30f)

// ---------------------------------------------------------------------------
// Kernel 0a: (B,C,L) f32 -> (B,L,C) f16 pack/transpose for query and key inputs
// ---------------------------------------------------------------------------
__global__ __launch_bounds__(256) void k_pack(const float* __restrict__ q,
                                              const float* __restrict__ k,
                                              _Float16* __restrict__ xq,
                                              _Float16* __restrict__ xk) {
    int idx = blockIdx.x * blockDim.x + threadIdx.x;
    const int total = BB * LL * CIN;
    if (idx >= total) return;
    int c = idx % CIN;
    int l = (idx / CIN) % LL;
    int b = idx / (CIN * LL);
    size_t src = ((size_t)b * CIN + c) * LL + l;
    xq[idx] = (_Float16)q[src];
    xk[idx] = (_Float16)k[src];
}

// ---------------------------------------------------------------------------
// Kernel 0b: convert Wq/Wk/Wv (256x192 f32, row-major) to f16 once.
// Wh layout: [pz][e][c], pz in {0:q,1:k,2:v}
// ---------------------------------------------------------------------------
__global__ __launch_bounds__(256) void k_packw(const float* __restrict__ Wq,
                                               const float* __restrict__ Wk,
                                               const float* __restrict__ Wv,
                                               _Float16* __restrict__ Wh) {
    int idx = blockIdx.x * blockDim.x + threadIdx.x;
    const int per = EMB * CIN;            // 49152
    if (idx >= 3 * per) return;
    int pz = idx / per;
    int r  = idx - pz * per;
    const float* W = (pz == 0) ? Wq : ((pz == 1) ? Wk : Wv);
    Wh[idx] = (_Float16)W[r];
}

// ---------------------------------------------------------------------------
// Kernel 1: projections via WMMA.
//   out[l][e] = bias[e] + sum_c x[l][c] * W[e][c]
// blockIdx.z: 0 -> Q (scaled by 1/sqrt(32), layout (B,h,L,32))
//             1 -> K (layout (B,h,L,32))
//             2 -> V (layout transposed (B,h,32,L))
// 8 waves/block, each wave owns 16 rows (l), full E=256.
// ---------------------------------------------------------------------------
__global__ __launch_bounds__(256) void k_proj(const _Float16* __restrict__ xq,
                                              const _Float16* __restrict__ xk,
                                              const _Float16* __restrict__ Wh,
                                              const float* __restrict__ bq,
                                              const float* __restrict__ bk,
                                              const float* __restrict__ bv,
                                              _Float16* __restrict__ Qo,
                                              _Float16* __restrict__ Ko,
                                              _Float16* __restrict__ Vt) {
    const int pz   = blockIdx.z;
    const int b    = blockIdx.y;
    const int wave = threadIdx.x >> 5;
    const int lane = threadIdx.x & 31;
    const int lh   = lane >> 4;   // lane half (0/1)
    const int ln   = lane & 15;
    const int r0   = blockIdx.x * 128 + wave * 16;   // first l-row of this wave

    const _Float16* X    = (pz == 0) ? xq : xk;
    const _Float16* W    = Wh + (size_t)pz * EMB * CIN;
    const float*    bias = (pz == 0) ? bq : ((pz == 1) ? bk : bv);
    const float     scale = (pz == 0) ? 0.17677669529663687f : 1.0f; // 1/sqrt(32)

    // A fragments: 6 chunks of K=32 over C=192.
    // A k order used here (contiguous, B matches): lane<16 holds K {0..15}, lane>=16 holds {16..31}.
    v16h a[6];
    const int arow = r0 + ln;
#pragma unroll
    for (int cc = 0; cc < 6; ++cc) {
        const _Float16* base = X + ((size_t)b * LL + arow) * CIN + cc * 32 + lh * 16;
        v8h lo = *(const v8h*)(base);
        v8h hi = *(const v8h*)(base + 8);
        v16h t;
#pragma unroll
        for (int i = 0; i < 8; ++i) { t[i] = lo[i]; t[i + 8] = hi[i]; }
        a[cc] = t;
    }

#pragma unroll 1
    for (int et = 0; et < 16; ++et) {
        const int e0 = et * 16;
        const float cv = bias[e0 + ln];
        v8f acc = {cv, cv, cv, cv, cv, cv, cv, cv};
#pragma unroll
        for (int cc = 0; cc < 6; ++cc) {
            // B[k][n] = W[e0+n][cc*32+k]; lane: n=ln, k = lh*16 .. lh*16+15
            v16h bf = *(const v16h*)(W + (size_t)(e0 + ln) * CIN + cc * 32 + lh * 16);
            acc = __builtin_amdgcn_wmma_f32_16x16x32_f16(
                false, a[cc], false, bf, (short)0, acc, false, false);
        }
        const int e    = e0 + ln;
        const int head = e >> 5;
        const int d    = e & 31;
        if (pz == 2) {
            // V transposed: Vt[(b*8+head)*32 + d][row], rows contiguous per lane
            v8h pk;
#pragma unroll
            for (int r = 0; r < 8; ++r) pk[r] = (_Float16)acc[r];
            _Float16* vp = Vt + (((size_t)(b * NH + head)) * HD + d) * LL + r0 + lh * 8;
            *(v8h*)vp = pk;
        } else {
            _Float16* dst = (pz == 0) ? Qo : Ko;
#pragma unroll
            for (int r = 0; r < 8; ++r) {
                const int row = r0 + lh * 8 + r;
                dst[(((size_t)(b * NH + head)) * LL + row) * HD + d] =
                    (_Float16)(acc[r] * scale);
            }
        }
    }
}

// ---------------------------------------------------------------------------
// Kernel 2: causal flash attention, one (b,head) per blockIdx.y,
// 8 independent waves per block, each wave owns 16 query rows.
// Online softmax over key chunks of 32; 2 QK + 2 PV WMMAs per chunk.
// ---------------------------------------------------------------------------
__global__ __launch_bounds__(256) void k_attn(const _Float16* __restrict__ Q,
                                              const _Float16* __restrict__ K,
                                              const _Float16* __restrict__ Vt,
                                              float* __restrict__ out) {
    __shared__ _Float16 ldsP[8][16 * 32];   // per-wave P tile (1 KB each)

    const int wave = threadIdx.x >> 5;
    const int lane = threadIdx.x & 31;
    const int lh   = lane >> 4;
    const int ln   = lane & 15;
    const int bh   = blockIdx.y;                   // b*8 + head
    const int r0   = blockIdx.x * 128 + wave * 16; // first query row

    const _Float16* Qb = Q  + (size_t)bh * LL * HD;
    const _Float16* Kb = K  + (size_t)bh * LL * HD;
    const _Float16* Vb = Vt + (size_t)bh * HD * LL;

    // Q A-fragment (already scaled by 1/sqrt(hd))
    v16h aq;
    {
        const _Float16* p = Qb + (size_t)(r0 + ln) * HD + lh * 16;
        v8h lo = *(const v8h*)p;
        v8h hi = *(const v8h*)(p + 8);
#pragma unroll
        for (int i = 0; i < 8; ++i) { aq[i] = lo[i]; aq[i + 8] = hi[i]; }
    }

    v8f acc0 = {}, acc1 = {};
    float mrow[8], lrow[8];
#pragma unroll
    for (int r = 0; r < 8; ++r) { mrow[r] = NEG_BIG; lrow[r] = 0.0f; }

    _Float16* P = &ldsP[wave][0];
    const int kend = r0 + 16;   // causal: keys needed only up to r0+15

#pragma unroll 1
    for (int j = 0; j < kend; j += 32) {
        // K B-fragments: B[k][n] = K[j + t*16 + n][k]
        v16h bk0 = *(const v16h*)(Kb + (size_t)(j + ln) * HD + lh * 16);
        v16h bk1 = *(const v16h*)(Kb + (size_t)(j + 16 + ln) * HD + lh * 16);

        v8f z = {};
        v8f s0 = __builtin_amdgcn_wmma_f32_16x16x32_f16(false, aq, false, bk0,
                                                        (short)0, z, false, false);
        v8f s1 = __builtin_amdgcn_wmma_f32_16x16x32_f16(false, aq, false, bk1,
                                                        (short)0, z, false, false);

        // causal mask + online softmax (row = r0 + lh*8 + r, col key = j+t*16+ln)
#pragma unroll
        for (int r = 0; r < 8; ++r) {
            const int qrow = r0 + lh * 8 + r;
            if (j + ln > qrow)      s0[r] = NEG_BIG;
            if (j + 16 + ln > qrow) s1[r] = NEG_BIG;

            float mx = fmaxf(s0[r], s1[r]);
#pragma unroll
            for (int off = 8; off >= 1; off >>= 1)
                mx = fmaxf(mx, __shfl_xor(mx, off, 32));
            const float mnew = fmaxf(mrow[r], mx);
            const float corr = __expf(mrow[r] - mnew);
            mrow[r] = mnew;

            const float p0 = __expf(s0[r] - mnew);
            const float p1 = __expf(s1[r] - mnew);
            float ps = p0 + p1;
#pragma unroll
            for (int off = 8; off >= 1; off >>= 1)
                ps += __shfl_xor(ps, off, 32);
            lrow[r] = lrow[r] * corr + ps;
            acc0[r] *= corr;
            acc1[r] *= corr;
            s0[r] = p0;
            s1[r] = p1;
        }

        // P (C-fragment layout) -> LDS row-major 16x32
#pragma unroll
        for (int r = 0; r < 8; ++r) {
            P[(lh * 8 + r) * 32 + ln]      = (_Float16)s0[r];
            P[(lh * 8 + r) * 32 + 16 + ln] = (_Float16)s1[r];
        }
        asm volatile("s_wait_dscnt 0" ::: "memory");

        // Reload P as A-fragment
        v16h ap;
        {
            const _Float16* pp = P + ln * 32 + lh * 16;
            v8h lo = *(const v8h*)pp;
            v8h hi = *(const v8h*)(pp + 8);
#pragma unroll
            for (int i = 0; i < 8; ++i) { ap[i] = lo[i]; ap[i + 8] = hi[i]; }
        }

        // V B-fragments from transposed V: B[kk][n] = Vt[t*16+n][j+kk]
        v16h bv0 = *(const v16h*)(Vb + (size_t)(ln) * LL + j + lh * 16);
        v16h bv1 = *(const v16h*)(Vb + (size_t)(16 + ln) * LL + j + lh * 16);

        acc0 = __builtin_amdgcn_wmma_f32_16x16x32_f16(false, ap, false, bv0,
                                                      (short)0, acc0, false, false);
        acc1 = __builtin_amdgcn_wmma_f32_16x16x32_f16(false, ap, false, bv1,
                                                      (short)0, acc1, false, false);
    }

    // Normalize and store: out[(b*256 + head*32 + d)][l], rows contiguous per lane
#pragma unroll
    for (int r = 0; r < 8; ++r) {
        const float inv = 1.0f / lrow[r];
        acc0[r] *= inv;
        acc1[r] *= inv;
    }
    const int b = bh >> 3, h = bh & 7;
    float* o0 = out + ((size_t)(b * EMB + h * HD + ln)) * LL + r0 + lh * 8;
    float* o1 = out + ((size_t)(b * EMB + h * HD + 16 + ln)) * LL + r0 + lh * 8;
    *(v8f*)o0 = acc0;
    *(v8f*)o1 = acc1;
}

// ---------------------------------------------------------------------------
extern "C" void kernel_launch(void* const* d_in, const int* in_sizes, int n_in,
                              void* d_out, int out_size, void* d_ws, size_t ws_size,
                              hipStream_t stream) {
    (void)in_sizes; (void)n_in; (void)out_size; (void)ws_size;

    const float* query = (const float*)d_in[0];
    const float* key   = (const float*)d_in[1];
    const float* Wq    = (const float*)d_in[2];
    const float* bq    = (const float*)d_in[3];
    const float* Wk    = (const float*)d_in[4];
    const float* bk    = (const float*)d_in[5];
    const float* Wv    = (const float*)d_in[6];
    const float* bv    = (const float*)d_in[7];
    float* out         = (float*)d_out;

    // workspace carve-up (f16 halves)
    _Float16* ws   = (_Float16*)d_ws;
    const size_t nx = (size_t)BB * LL * CIN;       // 884,736
    const size_t nq = (size_t)BB * NH * LL * HD;   // 1,179,648
    const size_t nw = (size_t)3 * EMB * CIN;       // 147,456
    _Float16* xq16 = ws;
    _Float16* xk16 = xq16 + nx;
    _Float16* Qh   = xk16 + nx;
    _Float16* Kh   = Qh + nq;
    _Float16* Vth  = Kh + nq;
    _Float16* Wh   = Vth + nq;
    (void)nw;

    const int total = BB * LL * CIN;
    k_pack<<<dim3((total + 255) / 256), 256, 0, stream>>>(query, key, xq16, xk16);
    k_packw<<<dim3((3 * EMB * CIN + 255) / 256), 256, 0, stream>>>(Wq, Wk, Wv, Wh);

    // 18 l-tiles of 128 rows, B=2, 3 projections
    k_proj<<<dim3(LL / 128, BB, 3), 256, 0, stream>>>(xq16, xk16, Wh, bq, bk, bv,
                                                      Qh, Kh, Vth);

    // 18 q-tiles x 16 (b,head) pairs
    k_attn<<<dim3(LL / 128, BB * NH), 256, 0, stream>>>(Qh, Kh, Vth, out);
}
